// GAT_H_7078106104033
// MI455X (gfx1250) — matrix operand order repaired and checked by reading, not compiled, not run on hardware
//
#include <hip/hip_runtime.h>
#include <hip/hip_bf16.h>
#include <math.h>

// Problem constants (must match reference)
#define NN   50000
#define EE   1200000
#define EP   (EE + NN)     // edges including self-loops
#define DIN  128
#define HH   4
#define CC   64
#define HCC  256
#define GG   8
#define OUTC 4

typedef __attribute__((ext_vector_type(16))) __bf16 v16bf;
typedef __attribute__((ext_vector_type(8)))  float  v8f;

// ---------- helpers ----------
// Order-preserving float<->int mapping (involution) so atomicMax(int) == max(float)
__device__ __forceinline__ int f2o(float f) {
    int i = __float_as_int(f);
    return (i >= 0) ? i : (i ^ 0x7fffffff);
}
__device__ __forceinline__ float o2f(int i) {
    int j = (i >= 0) ? i : (i ^ 0x7fffffff);
    return __int_as_float(j);
}

__global__ void fill_f32(float* __restrict__ p, float v, long n) {
    long t = (long)blockIdx.x * blockDim.x + threadIdx.x;
    if (t < n) p[t] = v;
}
__global__ void fill_i32(int* __restrict__ p, int v, long n) {
    long t = (long)blockIdx.x * blockDim.x + threadIdx.x;
    if (t < n) p[t] = v;
}

// ---------- GEMM: Hout[N,256] = Xin[N,K] @ W[K,256]  (bf16 WMMA, f32 acc) ----------
// Block tile 128x64; 8 waves; wave w = M-subtile w (16 rows) x 4 N-subtiles
// (4 accumulators -> 4 wmma per k-step, A-fragment reused 4x).
// Tiles are staged in LDS in FRAGMENT-MAJOR layout: writers swizzle each value to
// its (lane, element) WMMA-fragment position, so each lane's fragment is one
// contiguous 32B-aligned v16bf LDS read (wide ds_load instead of 16 scalar loads).
__global__ void __launch_bounds__(256)
gemm_bf16_wmma(const float* __restrict__ X, const float* __restrict__ W,
               float* __restrict__ Hout, int K, int relu_in)
{
    __shared__ v16bf aFrag[8 * 32];   // 8 M-subtiles x 32 lanes  (128x32 A tile, 8KB)
    __shared__ v16bf bFrag[4 * 32];   // 4 N-subtiles x 32 lanes  (32x64  B tile, 4KB)
    __bf16* const ap = reinterpret_cast<__bf16*>(aFrag);
    __bf16* const bp = reinterpret_cast<__bf16*>(bFrag);

    const int tid  = threadIdx.x;
    const int wave = tid >> 5;
    const int lane = tid & 31;
    const int rb   = (blockIdx.x >> 2) * 128;      // row-block base
    const int cb   = (blockIdx.x & 3) * 64;        // col-block base
    const int m    = lane & 15;

    v8f acc[4] = {};

    for (int k0 = 0; k0 < K; k0 += 32) {
        __syncthreads();   // previous iteration's fragment reads done

        // ---- A tile: 128x32 floats, coalesced float4, swizzled to fragment layout ----
        #pragma unroll
        for (int i = 0; i < 4; ++i) {
            int flat4 = tid + i * 256;             // float4 index in tile
            int row = flat4 >> 3;                  // 8 float4 per row (32 floats)
            int kq  = (flat4 & 7) * 4;
            int grow = rb + row; if (grow >= NN) grow = NN - 1;   // clamp tail
            const float4 v = *reinterpret_cast<const float4*>(
                X + (long)grow * K + k0 + kq);
            float xv[4] = { v.x, v.y, v.z, v.w };
            int mt = row >> 4, mm = row & 15;
            #pragma unroll
            for (int j = 0; j < 4; ++j) {
                int k  = kq + j;
                int nh = (k & 15) >> 3;                         // lane half
                int vg = ((k >> 4) << 2) + ((k & 7) >> 1);      // VGPR index
                int e  = (vg << 1) | (k & 1);                   // fragment element
                int ln = (nh << 4) | mm;
                float f = xv[j];
                if (relu_in) f = fmaxf(f, 0.f);
                ap[((mt * 32 + ln) << 4) + e] = (__bf16)f;
            }
        }
        // ---- B tile: 32x64 floats, coalesced float4, swizzled to fragment layout ----
        #pragma unroll
        for (int i = 0; i < 2; ++i) {
            int flat4 = tid + i * 256;
            int krow = flat4 >> 4;                 // 16 float4 per row (64 floats)
            int c4   = (flat4 & 15) * 4;
            const float4 v = *reinterpret_cast<const float4*>(
                W + (long)(k0 + krow) * HCC + cb + c4);
            float wv[4] = { v.x, v.y, v.z, v.w };
            int kh = (krow >> 3) & 1, e = ((krow >> 4) << 3) | (krow & 7);
            int nt = c4 >> 4;                      // 4-aligned c4 never crosses subtile
            #pragma unroll
            for (int j = 0; j < 4; ++j) {
                int n  = (c4 & 15) + j;
                int ln = (kh << 4) | n;
                bp[((nt * 32 + ln) << 4) + e] = (__bf16)wv[j];
            }
        }
        __syncthreads();

        // ---- fragments are contiguous per lane: single wide LDS read each ----
        const v16bf a = aFrag[wave * 32 + lane];
        #pragma unroll
        for (int nt = 0; nt < 4; ++nt) {
            const v16bf b = bFrag[nt * 32 + lane];
            acc[nt] = __builtin_amdgcn_wmma_f32_16x16x32_bf16(
                false, a, false, b, (short)0, acc[nt], false, false);
        }
    }

    // ---- epilogue: C/D layout VGPR r -> M = r + 8*(lane>=16); N = lane%16 ----
    const int rbase = rb + wave * 16 + ((lane >= 16) ? 8 : 0);
    #pragma unroll
    for (int nt = 0; nt < 4; ++nt) {
        const int col = cb + nt * 16 + m;
        #pragma unroll
        for (int r = 0; r < 8; ++r) {
            int row = rbase + r;
            if (row < NN) Hout[(long)row * HCC + col] = acc[nt][r];
        }
    }
}

// ---------- attention scores: asrc/adst [N,H] = sum_c h[n,h,c] * a_{s,d}[h,c] ----------
__global__ void alpha_kernel(const float* __restrict__ Hmat,
                             const float* __restrict__ as_w,
                             const float* __restrict__ ad_w,
                             float* __restrict__ asrc, float* __restrict__ adst)
{
    int t = blockIdx.x * blockDim.x + threadIdx.x;
    if (t >= NN * HH) return;
    int n = t >> 2, h = t & 3;
    const float* hp = Hmat + (long)n * HCC + h * CC;
    const float* aw = as_w + h * CC;
    const float* dw = ad_w + h * CC;
    float s0 = 0.f, s1 = 0.f;
    #pragma unroll 4
    for (int c = 0; c < CC; ++c) { float v = hp[c]; s0 += v * aw[c]; s1 += v * dw[c]; }
    asrc[t] = s0; adst[t] = s1;
}

__device__ __forceinline__ void edge_sd(const int* __restrict__ ei, int e, int& s, int& d) {
    if (e < EE) { s = ei[e]; d = ei[EE + e]; } else { s = e - EE; d = s; }
}
__device__ __forceinline__ float leaky(float v) { return (v > 0.f) ? v : 0.2f * v; }

// Pass A: segment max of leaky_relu(e) over dst
__global__ void edge_max(const int* __restrict__ ei, const float* __restrict__ asrc,
                         const float* __restrict__ adst, int* __restrict__ mbuf)
{
    long t = (long)blockIdx.x * blockDim.x + threadIdx.x;
    if (t >= (long)EP * HH) return;
    int e = (int)(t >> 2), h = (int)(t & 3);
    int s, d; edge_sd(ei, e, s, d);
    float v = leaky(asrc[s * 4 + h] + adst[d * 4 + h]);
    atomicMax(&mbuf[d * 4 + h], f2o(v));
}

// Pass B: segment sum of exp(e - m[dst])
__global__ void edge_sum(const int* __restrict__ ei, const float* __restrict__ asrc,
                         const float* __restrict__ adst, const int* __restrict__ mbuf,
                         float* __restrict__ sbuf)
{
    long t = (long)blockIdx.x * blockDim.x + threadIdx.x;
    if (t >= (long)EP * HH) return;
    int e = (int)(t >> 2), h = (int)(t & 3);
    int s, d; edge_sd(ei, e, s, d);
    int idx = d * 4 + h;
    float v = leaky(asrc[s * 4 + h] + adst[idx]);
    atomicAdd(&sbuf[idx], __expf(v - o2f(mbuf[idx])));
}

// Pass C: agg[dst] += h[src] * alpha   (1 thread per (edge, 4-channel chunk), head aligned)
__global__ void edge_aggregate(const int* __restrict__ ei, const float* __restrict__ asrc,
                               const float* __restrict__ adst, const int* __restrict__ mbuf,
                               const float* __restrict__ sbuf, const float* __restrict__ Hmat,
                               float* __restrict__ agg)
{
    long t = (long)blockIdx.x * blockDim.x + threadIdx.x;
    if (t >= (long)EP * 64) return;
    int e = (int)(t >> 6), part = (int)(t & 63);
    int h = part >> 4;                       // 16 chunks of 4 channels per head
    int s, d; edge_sd(ei, e, s, d);
    int idx = d * 4 + h;
    float v  = leaky(asrc[s * 4 + h] + adst[idx]);
    float ex = __expf(v - o2f(mbuf[idx]));
    float al = ex / (sbuf[idx] + 1e-16f);
    int cb = h * CC + (part & 15) * 4;
    const float4 hv = *reinterpret_cast<const float4*>(Hmat + (long)s * HCC + cb);
    float* op = agg + (long)d * HCC + cb;
    atomicAdd(op + 0, hv.x * al);
    atomicAdd(op + 1, hv.y * al);
    atomicAdd(op + 2, hv.z * al);
    atomicAdd(op + 3, hv.w * al);
}

// ---------- BatchNorm (training-style batch stats) ----------
__global__ void bn_stats(const float* __restrict__ agg, const float* __restrict__ bias,
                         float* __restrict__ sums, float* __restrict__ sqs)
{
    int ch = threadIdx.x;                     // 256 channels
    int r0 = blockIdx.x * 128;
    float s = 0.f, q = 0.f;
    for (int r = 0; r < 128; ++r) {
        int n = r0 + r;
        if (n < NN) {
            float v = agg[(long)n * HCC + ch] + bias[ch];
            s += v; q += v * v;
        }
    }
    atomicAdd(&sums[ch], s);
    atomicAdd(&sqs[ch], q);
}

__global__ void bn_apply_relu(const float* __restrict__ agg, const float* __restrict__ bias,
                              const float* __restrict__ sums, const float* __restrict__ sqs,
                              const float* __restrict__ gam, const float* __restrict__ bet,
                              float* __restrict__ out)
{
    long t = (long)blockIdx.x * blockDim.x + threadIdx.x;
    if (t >= (long)NN * HCC) return;
    int ch = (int)(t & 255);
    float mu  = sums[ch] * (1.0f / NN);
    float var = sqs[ch] * (1.0f / NN) - mu * mu;
    float v = agg[t] + bias[ch];
    float y = (v - mu) * rsqrtf(var + 1e-5f) * gam[ch] + bet[ch];
    out[t] = fmaxf(y, 0.0f);
}

// ---------- global mean pool ----------
__global__ void count_kernel(const int* __restrict__ batch, float* __restrict__ cnt) {
    int t = blockIdx.x * blockDim.x + threadIdx.x;
    if (t < NN) atomicAdd(&cnt[batch[t]], 1.0f);
}
__global__ void pool_kernel(const float* __restrict__ agg, const float* __restrict__ bias,
                            const int* __restrict__ batch, float* __restrict__ pooled)
{
    long t = (long)blockIdx.x * blockDim.x + threadIdx.x;
    if (t >= (long)NN * 64) return;
    int n = (int)(t >> 6), part = (int)(t & 63);
    int g = batch[n];
    int cb = part * 4;
    const float4 v = *reinterpret_cast<const float4*>(agg + (long)n * HCC + cb);
    const float4 b = *reinterpret_cast<const float4*>(bias + cb);
    float* op = pooled + g * HCC + cb;
    atomicAdd(op + 0, v.x + b.x);
    atomicAdd(op + 1, v.y + b.y);
    atomicAdd(op + 2, v.z + b.z);
    atomicAdd(op + 3, v.w + b.w);
}

// ---------- MLP head + softmax + argmax (single block, 64 threads) ----------
__global__ void head_kernel(const float* __restrict__ pooled, const float* __restrict__ cnt,
                            const float* __restrict__ fc1w, const float* __restrict__ fc1b,
                            const float* __restrict__ fc2w, const float* __restrict__ fc2b,
                            float* __restrict__ out)
{
    __shared__ float pm[GG * HCC];
    __shared__ float hid[GG * CC];
    int tid = threadIdx.x;
    for (int i = tid; i < GG * HCC; i += 64) {
        int g = i >> 8;
        pm[i] = pooled[i] / fmaxf(cnt[g], 1.0f);
    }
    __syncthreads();
    int j = tid;                               // 64 threads = 64 hidden units
    for (int g = 0; g < GG; ++g) {
        float s = fc1b[j];
        for (int c = 0; c < HCC; ++c) s += pm[g * HCC + c] * fc1w[c * CC + j];
        hid[g * CC + j] = s;
    }
    __syncthreads();
    if (tid < GG) {
        int g = tid;
        float l[OUTC];
        #pragma unroll
        for (int o = 0; o < OUTC; ++o) {
            float s = fc2b[o];
            for (int jj = 0; jj < CC; ++jj) s += hid[g * CC + jj] * fc2w[jj * OUTC + o];
            l[o] = s;
        }
        float mx = l[0]; int am = 0;
        #pragma unroll
        for (int o = 1; o < OUTC; ++o) if (l[o] > mx) { mx = l[o]; am = o; }
        float es = 0.f, p[OUTC];
        #pragma unroll
        for (int o = 0; o < OUTC; ++o) { p[o] = __expf(l[o] - mx); es += p[o]; }
        #pragma unroll
        for (int o = 0; o < OUTC; ++o) {
            out[g * OUTC + o] = l[o];                     // logits
            out[GG * OUTC + g * OUTC + o] = p[o] / es;    // probs
        }
        out[2 * GG * OUTC + g] = (float)am;               // argmax
    }
}

// ---------- host-side per-layer sequence ----------
static void run_gat_layer(const float* Xin, int K, int relu_in,
                          const float* W, const float* as_w, const float* ad_w,
                          float* hbuf, float* asrc, float* adst,
                          int* mbuf, float* sbuf, float* agg,
                          const int* ei, hipStream_t stream)
{
    const int NEG_INF_ORD = (int)0x807FFFFF;   // f2o(-inf)
    const int gemm_blocks = ((NN + 127) / 128) * 4;   // 391 row-blocks x 4 col-blocks
    gemm_bf16_wmma<<<gemm_blocks, 256, 0, stream>>>(Xin, W, hbuf, K, relu_in);
    alpha_kernel<<<(NN * HH + 255) / 256, 256, 0, stream>>>(hbuf, as_w, ad_w, asrc, adst);
    fill_i32<<<(NN * HH + 255) / 256, 256, 0, stream>>>(mbuf, NEG_INF_ORD, (long)NN * HH);
    fill_f32<<<(NN * HH + 255) / 256, 256, 0, stream>>>(sbuf, 0.0f, (long)NN * HH);
    fill_f32<<<(int)(((long)NN * HCC + 255) / 256), 256, 0, stream>>>(agg, 0.0f, (long)NN * HCC);
    edge_max<<<(int)(((long)EP * HH + 255) / 256), 256, 0, stream>>>(ei, asrc, adst, mbuf);
    edge_sum<<<(int)(((long)EP * HH + 255) / 256), 256, 0, stream>>>(ei, asrc, adst, mbuf, sbuf);
    edge_aggregate<<<(int)(((long)EP * 64 + 255) / 256), 256, 0, stream>>>(
        ei, asrc, adst, mbuf, sbuf, hbuf, agg);
}

extern "C" void kernel_launch(void* const* d_in, const int* in_sizes, int n_in,
                              void* d_out, int out_size, void* d_ws, size_t ws_size,
                              hipStream_t stream)
{
    const float* x     = (const float*)d_in[0];
    const int*   ei    = (const int*)  d_in[1];
    const int*   batch = (const int*)  d_in[2];
    const float* W0  = (const float*)d_in[3];
    const float* as0 = (const float*)d_in[4];
    const float* ad0 = (const float*)d_in[5];
    const float* b0  = (const float*)d_in[6];
    const float* W1  = (const float*)d_in[7];
    const float* as1 = (const float*)d_in[8];
    const float* ad1 = (const float*)d_in[9];
    const float* b1  = (const float*)d_in[10];
    const float* W2  = (const float*)d_in[11];
    const float* as2 = (const float*)d_in[12];
    const float* ad2 = (const float*)d_in[13];
    const float* b2  = (const float*)d_in[14];
    const float* g0  = (const float*)d_in[15];
    const float* be0 = (const float*)d_in[16];
    const float* g1  = (const float*)d_in[17];
    const float* be1 = (const float*)d_in[18];
    const float* fc1w = (const float*)d_in[19];
    const float* fc1b = (const float*)d_in[20];
    const float* fc2w = (const float*)d_in[21];
    const float* fc2b = (const float*)d_in[22];

    // workspace layout (floats)
    float* base  = (float*)d_ws;
    float* xbuf  = base;                               // N*256  (BN output / next-layer input)
    float* hbuf  = xbuf + (size_t)NN * HCC;            // N*256  (GEMM output)
    float* agg   = hbuf + (size_t)NN * HCC;            // N*256  (attention aggregate)
    float* asrc  = agg  + (size_t)NN * HCC;            // N*4
    float* adst  = asrc + (size_t)NN * HH;             // N*4
    float* sbuf  = adst + (size_t)NN * HH;             // N*4
    int*   mbuf  = (int*)(sbuf + (size_t)NN * HH);     // N*4 ints
    float* bns   = (float*)(mbuf + (size_t)NN * HH);   // 256
    float* bnq   = bns + HCC;                          // 256
    float* pooled= bnq + HCC;                          // 8*256
    float* cnt   = pooled + (size_t)GG * HCC;          // 8

    const int bn_blocks    = (NN + 127) / 128;
    const int apply_blocks = (int)(((long)NN * HCC + 255) / 256);

    // ----- layer 0 (ReLU on input fused into GEMM A-load) -----
    run_gat_layer(x, DIN, 1, W0, as0, ad0, hbuf, asrc, adst, mbuf, sbuf, agg, ei, stream);
    fill_f32<<<2, 256, 0, stream>>>(bns, 0.0f, 2 * HCC);   // bns + bnq contiguous
    bn_stats<<<bn_blocks, 256, 0, stream>>>(agg, b0, bns, bnq);
    bn_apply_relu<<<apply_blocks, 256, 0, stream>>>(agg, b0, bns, bnq, g0, be0, xbuf);

    // ----- layer 1 -----
    run_gat_layer(xbuf, HCC, 0, W1, as1, ad1, hbuf, asrc, adst, mbuf, sbuf, agg, ei, stream);
    fill_f32<<<2, 256, 0, stream>>>(bns, 0.0f, 2 * HCC);
    bn_stats<<<bn_blocks, 256, 0, stream>>>(agg, b1, bns, bnq);
    bn_apply_relu<<<apply_blocks, 256, 0, stream>>>(agg, b1, bns, bnq, g1, be1, xbuf);

    // ----- layer 2 (no BN; bias b2 added during pooling) -----
    run_gat_layer(xbuf, HCC, 0, W2, as2, ad2, hbuf, asrc, adst, mbuf, sbuf, agg, ei, stream);

    // ----- pool + head -----
    fill_f32<<<9, 256, 0, stream>>>(pooled, 0.0f, (long)GG * HCC + GG); // pooled + cnt contiguous
    count_kernel<<<(NN + 255) / 256, 256, 0, stream>>>(batch, cnt);
    pool_kernel<<<(int)(((long)NN * 64 + 255) / 256), 256, 0, stream>>>(agg, b2, batch, pooled);
    head_kernel<<<1, 64, 0, stream>>>(pooled, cnt, fc1w, fc1b, fc2w, fc2b, (float*)d_out);

    (void)in_sizes; (void)n_in; (void)out_size; (void)ws_size;
}
